// DigitCapsv2_10_77816217469391
// MI455X (gfx1250) — matrix operand order repaired and checked by reading, not compiled, not run on hardware
//
#include <hip/hip_runtime.h>
#include <hip/hip_bf16.h>
#include <math.h>

// ---------------------------------------------------------------------------
// DigitCaps dynamic routing for MI455X (gfx1250, wave32, WMMA).
// All heavy matmuls use v_wmma_f32_16x16x32_bf16 with 32x32 wave tiles
// (4 accumulators: 2 A-tiles x 2 B-tiles) for 2:1 vmem:wmma issue ratio.
// Workspace requirement: ~126 MB (see layout in kernel_launch).
// ---------------------------------------------------------------------------

typedef __attribute__((ext_vector_type(16))) __bf16 v16bf;
typedef __attribute__((ext_vector_type(8)))  __bf16 v8bf;
typedef __attribute__((ext_vector_type(8)))  float  v8f;

#define NB   10      // batch
#define NS   50      // sequence / routing columns
#define ND   3072    // feature dim
#define NP   768     // capsule rows
#define NKS  160     // padded K for the s-GEMM (3*50 -> 160)
#define TROWS 192    // padded rows of stacked T (150 -> 192, pad for n<=63 reads)
#define NR   2304    // 3*768 softmax axis

__device__ __forceinline__ v16bf pack16(v8bf lo, v8bf hi) {
    union { v16bf v; v8bf h[2]; } u;
    u.h[0] = lo; u.h[1] = hi;
    return u.v;
}

// A-operand 16x32 bf16 tile (row-major source, ld in elements).
// Lane l: m = m0 + (l&15); half = l>>4; elems 0..7 = K[k0+8h .. +7],
// elems 8..15 = K[k0+16+8h .. +7].  (ISA 7.12.2, 16-bit A matrix)
__device__ __forceinline__ v16bf load_a16(const __bf16* A, int lda, int m0, int k0, int lane) {
    int m = m0 + (lane & 15);
    int half = lane >> 4;
    const __bf16* p = A + (size_t)m * lda + k0 + 8 * half;
    v8bf lo = *(const v8bf*)p;
    v8bf hi = *(const v8bf*)(p + 16);
    return pack16(lo, hi);
}

// B-operand 32x16 bf16 tile from COLUMN-major storage Bc[n][k] (ld in elems).
// Lane l: n = n0 + (l&15); half = l>>4; elems 0..7 = K[k0+8h .. +7], elems 8..15 = K[k0+16+8h .. +7].
__device__ __forceinline__ v16bf load_b16(const __bf16* Bc, int ldb, int n0, int k0, int lane) {
    int n = n0 + (lane & 15);
    int half = lane >> 4;
    const __bf16* p = Bc + (size_t)n * ldb + k0 + 8 * half;
    v8bf lo = *(const v8bf*)p;
    v8bf hi = *(const v8bf*)(p + 16);
    return pack16(lo, hi);
}

__device__ __forceinline__ v8f wmma_bf16(v16bf a, v16bf b, v8f c) {
    return __builtin_amdgcn_wmma_f32_16x16x32_bf16(
        false, a, false, b, (short)0, c, false, false);
}

// ---------------------------------------------------------------------------
// Prep: stack visual/acoustic/va into T[b][192][3072] bf16 (rows 150..191 = 0)
// and its transpose Tt[b][3072][160] bf16 for the s-GEMM B operand.
// ---------------------------------------------------------------------------
__global__ __launch_bounds__(256) void prep_T(const float* __restrict__ vis,
                                              const float* __restrict__ aco,
                                              const float* __restrict__ va,
                                              __bf16* __restrict__ T,
                                              __bf16* __restrict__ Tt) {
    int r = blockIdx.x;          // 0..191
    int b = blockIdx.y;
    const float* src = nullptr;
    if (r < 50)       src = vis + ((size_t)b * NS + r)       * ND;
    else if (r < 100) src = aco + ((size_t)b * NS + (r-50))  * ND;
    else if (r < 150) src = va  + ((size_t)b * NS + (r-100)) * ND;
    for (int d = threadIdx.x; d < ND; d += 256) {
        float v = src ? src[d] : 0.0f;
        __bf16 h = (__bf16)v;
        T[((size_t)b * TROWS + r) * ND + d] = h;
        if (r < NKS)
            Tt[((size_t)b * ND + d) * NKS + r] = h;
    }
}

// W (10,50,768) -> Wb[b][64][768] bf16, rows 50..63 zero.
__global__ __launch_bounds__(256) void prep_W(const float* __restrict__ W,
                                              __bf16* __restrict__ Wb) {
    int i = blockIdx.x;          // 0..63
    int b = blockIdx.y;
    for (int k = threadIdx.x; k < NP; k += 256) {
        float v = (i < NS) ? W[((size_t)b * NS + i) * NP + k] : 0.0f;
        Wb[((size_t)b * 64 + i) * NP + k] = (__bf16)v;
    }
}

// ---------------------------------------------------------------------------
// Softmax over the 2304-row axis for one (batch, column j); emits routing
// coefficients C[b][m][mod*50+j] bf16 (A-operand layout for the s-GEMM).
// ---------------------------------------------------------------------------
__global__ __launch_bounds__(256) void softmax_cols(const float* __restrict__ blog,
                                                    __bf16* __restrict__ C) {
    int j = blockIdx.x;          // 0..49
    int b = blockIdx.y;
    const float* col = blog + (size_t)b * NR * 64 + j;
    __shared__ float red[256];
    int tid = threadIdx.x;

    float mx = -1e30f;
    for (int r = tid; r < NR; r += 256) mx = fmaxf(mx, col[(size_t)r * 64]);
    red[tid] = mx; __syncthreads();
    for (int s = 128; s > 0; s >>= 1) {
        if (tid < s) red[tid] = fmaxf(red[tid], red[tid + s]);
        __syncthreads();
    }
    float smax = red[0]; __syncthreads();

    float sum = 0.0f;
    for (int r = tid; r < NR; r += 256) sum += __expf(col[(size_t)r * 64] - smax);
    red[tid] = sum; __syncthreads();
    for (int s = 128; s > 0; s >>= 1) {
        if (tid < s) red[tid] += red[tid + s];
        __syncthreads();
    }
    float inv = 1.0f / red[0];

    for (int r = tid; r < NR; r += 256) {
        float v = __expf(col[(size_t)r * 64] - smax) * inv;
        int mod = r / NP, m = r % NP;
        C[((size_t)b * NP + m) * NKS + mod * NS + j] = (__bf16)v;
    }
}

// ---------------------------------------------------------------------------
// s-GEMM: S[b] = C[b](768x160) x T[b](160x3072); 32x32 wave tile.
// Store bf16 s into Abuf and accumulate per-row sum-of-squares into sq via
// lane-shuffle + atomicAdd.
// ---------------------------------------------------------------------------
__global__ __launch_bounds__(256) void gemm_s(const __bf16* __restrict__ C,
                                              const __bf16* __restrict__ Tt,
                                              __bf16* __restrict__ Abuf,
                                              float* __restrict__ sq) {
    int b = blockIdx.y;
    int wave = threadIdx.x >> 5, lane = threadIdx.x & 31;
    int patch = blockIdx.x * 8 + wave;           // 24 m-pairs * 96 n-pairs = 2304
    int mp = patch / 96, np = patch % 96;
    int m0 = mp * 32, n0 = np * 32;

    const __bf16* A  = C  + (size_t)b * NP * NKS;
    const __bf16* Bc = Tt + (size_t)b * ND * NKS;

    v8f acc00 = {}, acc01 = {}, acc10 = {}, acc11 = {};
#pragma unroll
    for (int k0 = 0; k0 < NKS; k0 += 32) {
        v16bf a0 = load_a16(A, NKS, m0,      k0, lane);
        v16bf a1 = load_a16(A, NKS, m0 + 16, k0, lane);
        v16bf b0 = load_b16(Bc, NKS, n0,      k0, lane);
        v16bf b1 = load_b16(Bc, NKS, n0 + 16, k0, lane);
        acc00 = wmma_bf16(a0, b0, acc00);
        acc01 = wmma_bf16(a0, b1, acc01);
        acc10 = wmma_bf16(a1, b0, acc10);
        acc11 = wmma_bf16(a1, b1, acc11);
    }

    int half = lane >> 4, nl = lane & 15;
    __bf16* orow = Abuf + (size_t)b * NP * ND;
#pragma unroll
    for (int v = 0; v < 8; ++v) {
        // rows m0 .. m0+15 (accumulators *0*)
        int mA = m0 + v + 8 * half;
        float x0 = acc00[v], x1 = acc01[v];
        orow[(size_t)mA * ND + n0 + nl]      = (__bf16)x0;
        orow[(size_t)mA * ND + n0 + 16 + nl] = (__bf16)x1;
        float ssA = x0 * x0 + x1 * x1;
        ssA += __shfl_xor(ssA, 1, 32);
        ssA += __shfl_xor(ssA, 2, 32);
        ssA += __shfl_xor(ssA, 4, 32);
        ssA += __shfl_xor(ssA, 8, 32);
        if (nl == 0) atomicAdd(&sq[(size_t)b * NP + mA], ssA);

        // rows m0+16 .. m0+31 (accumulators *1*)
        int mB = m0 + 16 + v + 8 * half;
        float y0 = acc10[v], y1 = acc11[v];
        orow[(size_t)mB * ND + n0 + nl]      = (__bf16)y0;
        orow[(size_t)mB * ND + n0 + 16 + nl] = (__bf16)y1;
        float ssB = y0 * y0 + y1 * y1;
        ssB += __shfl_xor(ssB, 1, 32);
        ssB += __shfl_xor(ssB, 2, 32);
        ssB += __shfl_xor(ssB, 4, 32);
        ssB += __shfl_xor(ssB, 8, 32);
        if (nl == 0) atomicAdd(&sq[(size_t)b * NP + mB], ssB);
    }
}

// ---------------------------------------------------------------------------
// Squash: scale each row of Abuf in place, also emit transposed copy aT
// (B operand for the final GEMM).
// ---------------------------------------------------------------------------
__global__ __launch_bounds__(256) void squash_rows(__bf16* __restrict__ Abuf,
                                                   __bf16* __restrict__ aT,
                                                   const float* __restrict__ sq) {
    int r = blockIdx.x;          // 0..7679 = b*768+m
    int b = r / NP, m = r % NP;
    float q = sq[r];
    float scale = (q / (1.0f + q)) / (sqrtf(q) + 1e-8f);
    __bf16* row = Abuf + (size_t)r * ND;
    for (int d = threadIdx.x; d < ND; d += 256) {
        float v = (float)row[d] * scale;
        __bf16 h = (__bf16)v;
        row[d] = h;
        aT[((size_t)b * ND + d) * NP + m] = h;
    }
}

// ---------------------------------------------------------------------------
// b-update GEMM: blog[b][mod*768+m][j] += a(768x3072) x T_mod^T(3072x50).
// B operand column j of T_mod^T == row (mod*50+j) of T (row-major, contig K).
// 32x32 wave tile; K loop = 3072/32 = 96 steps x 4 WMMAs.
// ---------------------------------------------------------------------------
__global__ __launch_bounds__(256) void gemm_bupd(const __bf16* __restrict__ Abuf,
                                                 const __bf16* __restrict__ T,
                                                 float* __restrict__ blog) {
    int b = blockIdx.y, mod = blockIdx.z;
    int wave = threadIdx.x >> 5, lane = threadIdx.x & 31;
    int tile = blockIdx.x * 8 + wave;            // 24 m-pairs * 2 n-pairs = 48
    int mp = tile / 2, np = tile % 2;
    int m0 = mp * 32, n0 = np * 32;

    const __bf16* A  = Abuf + (size_t)b * NP * ND;
    const __bf16* Bc = T + ((size_t)b * TROWS + mod * NS) * ND;   // rows -> columns

    v8f acc00 = {}, acc01 = {}, acc10 = {}, acc11 = {};
    for (int k0 = 0; k0 < ND; k0 += 32) {
        v16bf a0 = load_a16(A,  ND, m0,      k0, lane);
        v16bf a1 = load_a16(A,  ND, m0 + 16, k0, lane);
        v16bf b0 = load_b16(Bc, ND, n0,      k0, lane);
        v16bf b1 = load_b16(Bc, ND, n0 + 16, k0, lane);
        acc00 = wmma_bf16(a0, b0, acc00);
        acc01 = wmma_bf16(a0, b1, acc01);
        acc10 = wmma_bf16(a1, b0, acc10);
        acc11 = wmma_bf16(a1, b1, acc11);
    }

    int half = lane >> 4, nl = lane & 15;
    float* brow = blog + ((size_t)b * NR + (size_t)mod * NP) * 64;
#pragma unroll
    for (int v = 0; v < 8; ++v) {
        int mA = m0 + v + 8 * half;
        int mB = mA + 16;
        brow[(size_t)mA * 64 + n0 + nl]      += acc00[v];   // unique (m,n): no race
        brow[(size_t)mA * 64 + n0 + 16 + nl] += acc01[v];
        brow[(size_t)mB * 64 + n0 + nl]      += acc10[v];
        brow[(size_t)mB * 64 + n0 + 16 + nl] += acc11[v];
    }
}

// ---------------------------------------------------------------------------
// Final: out[b] = W(50x768) x a(768x3072), padded M to 64; B operand is aT.
// 32x32 wave tile.
// ---------------------------------------------------------------------------
__global__ __launch_bounds__(256) void gemm_final(const __bf16* __restrict__ Wb,
                                                  const __bf16* __restrict__ aT,
                                                  float* __restrict__ out) {
    int b = blockIdx.y;
    int wave = threadIdx.x >> 5, lane = threadIdx.x & 31;
    int patch = blockIdx.x * 8 + wave;           // 2 m-pairs * 96 n-pairs = 192
    int mp = patch / 96, np = patch % 96;
    int m0 = mp * 32, n0 = np * 32;

    const __bf16* A  = Wb + (size_t)b * 64 * NP;
    const __bf16* Bc = aT + (size_t)b * ND * NP;

    v8f acc00 = {}, acc01 = {}, acc10 = {}, acc11 = {};
    for (int k0 = 0; k0 < NP; k0 += 32) {
        v16bf a0 = load_a16(A, NP, m0,      k0, lane);
        v16bf a1 = load_a16(A, NP, m0 + 16, k0, lane);
        v16bf b0 = load_b16(Bc, NP, n0,      k0, lane);
        v16bf b1 = load_b16(Bc, NP, n0 + 16, k0, lane);
        acc00 = wmma_bf16(a0, b0, acc00);
        acc01 = wmma_bf16(a0, b1, acc01);
        acc10 = wmma_bf16(a1, b0, acc10);
        acc11 = wmma_bf16(a1, b1, acc11);
    }

    int half = lane >> 4, nl = lane & 15;
#pragma unroll
    for (int v = 0; v < 8; ++v) {
        int mA = m0 + v + 8 * half;
        if (mA < NS) {
            float* orow = out + ((size_t)b * NS + mA) * ND;
            orow[n0 + nl]      = acc00[v];
            orow[n0 + 16 + nl] = acc01[v];
        }
        int mB = m0 + 16 + v + 8 * half;
        if (mB < NS) {
            float* orow = out + ((size_t)b * NS + mB) * ND;
            orow[n0 + nl]      = acc10[v];
            orow[n0 + 16 + nl] = acc11[v];
        }
    }
}

// ---------------------------------------------------------------------------
extern "C" void kernel_launch(void* const* d_in, const int* in_sizes, int n_in,
                              void* d_out, int out_size, void* d_ws, size_t ws_size,
                              hipStream_t stream) {
    const float* vis = (const float*)d_in[0];
    const float* aco = (const float*)d_in[1];
    const float* va  = (const float*)d_in[2];
    const float* W   = (const float*)d_in[3];
    float* out = (float*)d_out;

    // ---- workspace layout (bytes), ~126 MB total -------------------------
    char* ws = (char*)d_ws;
    size_t off = 0;
    auto carve = [&](size_t bytes) { char* p = ws + off; off += (bytes + 255) & ~(size_t)255; return p; };
    __bf16* T    = (__bf16*)carve((size_t)NB * TROWS * ND * 2);   // 11.8 MB
    __bf16* Tt   = (__bf16*)carve((size_t)NB * ND * NKS * 2);     //  9.8 MB
    __bf16* Wb   = (__bf16*)carve((size_t)NB * 64 * NP * 2);      //  1.0 MB
    __bf16* C    = (__bf16*)carve((size_t)NB * NP * NKS * 2);     //  2.5 MB
    __bf16* Abuf = (__bf16*)carve((size_t)NB * NP * ND * 2);      // 47.2 MB
    __bf16* aT   = (__bf16*)carve((size_t)NB * ND * NP * 2);      // 47.2 MB
    float*  blog = (float*) carve((size_t)NB * NR * 64 * 4);      //  5.9 MB
    float*  sq   = (float*) carve((size_t)NB * NP * 4);           // 30 KB
    (void)ws_size;

    // ---- staging (bf16 conversion + transposes) --------------------------
    prep_T<<<dim3(TROWS, NB), 256, 0, stream>>>(vis, aco, va, T, Tt);
    prep_W<<<dim3(64, NB), 256, 0, stream>>>(W, Wb);
    hipMemsetAsync(blog, 0, (size_t)NB * NR * 64 * 4, stream);
    hipMemsetAsync(C,    0, (size_t)NB * NP * NKS * 2, stream);   // zero pad cols 150..159

    // ---- 3 routing iterations --------------------------------------------
    for (int it = 0; it < 3; ++it) {
        softmax_cols<<<dim3(NS, NB), 256, 0, stream>>>(blog, C);
        hipMemsetAsync(sq, 0, (size_t)NB * NP * 4, stream);
        gemm_s<<<dim3(288, NB), 256, 0, stream>>>(C, Tt, Abuf, sq);
        squash_rows<<<dim3(NB * NP), 256, 0, stream>>>(Abuf, aT, sq);
        if (it < 2)
            gemm_bupd<<<dim3(6, NB, 3), 256, 0, stream>>>(Abuf, T, blog);
    }

    // ---- out = W x a ------------------------------------------------------
    gemm_final<<<dim3(24, NB), 256, 0, stream>>>(Wb, aT, out);
    (void)in_sizes; (void)n_in; (void)out_size;
}
